// TreeSummarize_39109972197979
// MI455X (gfx1250) — matrix operand order repaired and checked by reading, not compiled, not run on hardware
//
#include <hip/hip_runtime.h>

// ---------------------------------------------------------------------------
// TreeLSTM (child-sum over topologically ordered parent pointers)
//   N = 4096 nodes, D_IN = 1024, D = 1024, 3D = 3072
// Phase 1: bf16 WMMA GEMMs for the batched input projections.
// Phase 2: persistent-grid sequential recurrence (GEMV per step, grid barriers).
// ---------------------------------------------------------------------------

#define TN      4096
#define TDIN    1024
#define TD      1024
#define TD3     3072
#define REC_NWG 32          // persistent workgroups in recurrence kernel

typedef __attribute__((ext_vector_type(16))) __bf16 v16bf;
typedef __attribute__((ext_vector_type(8)))  __bf16 v8bf;
typedef __attribute__((ext_vector_type(8)))  float  v8f;

// ------------------------------ conversions --------------------------------

__global__ void ts_f32_to_bf16(const float* __restrict__ src,
                               __bf16* __restrict__ dst, size_t n) {
  size_t i = (size_t)blockIdx.x * blockDim.x + threadIdx.x;
  if (i < n) dst[i] = (__bf16)src[i];
}

// src [rows, cols] f32 -> dst [cols, rows] bf16 (one-shot weight transpose)
__global__ void ts_f32_to_bf16_T(const float* __restrict__ src,
                                 __bf16* __restrict__ dst,
                                 int rows, int cols) {
  size_t i = (size_t)blockIdx.x * blockDim.x + threadIdx.x;
  size_t n = (size_t)rows * cols;
  if (i < n) {
    int r = (int)(i / cols);
    int c = (int)(i % cols);
    dst[(size_t)c * rows + r] = (__bf16)src[i];
  }
}

__global__ void ts_zero_f32(float* __restrict__ p, size_t n) {
  size_t i = (size_t)blockIdx.x * blockDim.x + threadIdx.x;
  if (i < n) p[i] = 0.0f;
}

// ------------------------------ WMMA GEMM ----------------------------------
// C[M,Nc] = A[M,K](bf16) * Bt[Nc,K](bf16, pre-transposed) + bias[Nc]
// Workgroup: 256 threads = 8 waves arranged 2(M) x 4(N); each wave owns a
// 16x16 C tile -> 32x64 tile per WG.  K in steps of 32 via
// v_wmma_f32_16x16x32_bf16.
//
// 16-bit A 16x32 fragment (ISA 7.12.2): lanes 0-15 -> row M=lane,
//   VGPR0-3 = K 0..7, VGPR4-7 = K 16..23; lanes 16-31 -> K 8..15 / 24..31.
// 16-bit B 32x16 fragment: lane 0-15 -> col N=lane, K 0..7 / 16..23;
//   lanes 16-31 -> K 8..15 / 24..31.  Both load as two 16-byte chunks per lane.

__global__ __launch_bounds__(256)
void ts_gemm_bf16_wmma(const __bf16* __restrict__ A,
                       const __bf16* __restrict__ Bt,
                       const float*  __restrict__ bias,
                       float* __restrict__ C,
                       int M, int Nc, int K) {
  const int tid  = threadIdx.x;
  const int lane = tid & 31;
  const int w    = tid >> 5;   // wave 0..7
  const int wm   = w >> 2;     // 0..1
  const int wn   = w & 3;      // 0..3
  const int sub  = lane & 15;
  const int hl   = lane >> 4;  // half-wave 0/1

  const int mBase = blockIdx.y * 32 + wm * 16;
  const int nBase = blockIdx.x * 64 + wn * 16;

  const __bf16* aRow = A  + (size_t)(mBase + sub) * K;
  const __bf16* bRow = Bt + (size_t)(nBase + sub) * K;

  v8f acc = {};
  for (int k0 = 0; k0 < K; k0 += 32) {
    const int ka = k0 + hl * 8;
    v8bf alo = *(const v8bf*)(aRow + ka);
    v8bf ahi = *(const v8bf*)(aRow + ka + 16);
    v16bf a;
#pragma unroll
    for (int i = 0; i < 8; ++i) { a[i] = alo[i]; a[i + 8] = ahi[i]; }

    const int kb = k0 + hl * 8;
    v8bf blo = *(const v8bf*)(bRow + kb);
    v8bf bhi = *(const v8bf*)(bRow + kb + 16);
    v16bf b;
#pragma unroll
    for (int i = 0; i < 8; ++i) { b[i] = blo[i]; b[i + 8] = bhi[i]; }

    acc = __builtin_amdgcn_wmma_f32_16x16x32_bf16(
        /*neg_a=*/false, a, /*neg_b=*/false, b,
        /*c_mod=*/(short)0, acc, /*reuse_a=*/false, /*reuse_b=*/false);
  }

  const int   col = nBase + sub;
  const float bv  = bias[col];
#pragma unroll
  for (int v = 0; v < 8; ++v) {
    int row = mBase + v + hl * 8;
    C[(size_t)row * Nc + col] = acc[v] + bv;
  }
}

// --------------------------- recurrence kernel -----------------------------

__device__ __forceinline__ float ts_sigmoid(float x) {
  return 1.0f / (1.0f + __expf(-x));
}

// Monotonic-counter device-wide barrier (REC_NWG workgroups, all resident).
__device__ __forceinline__ void ts_grid_barrier(unsigned* cnt, unsigned target) {
  __syncthreads();
  if (threadIdx.x == 0) {
    __threadfence();  // release our stores to device scope
    __hip_atomic_fetch_add(cnt, 1u, __ATOMIC_RELEASE, __HIP_MEMORY_SCOPE_AGENT);
    while (__hip_atomic_load(cnt, __ATOMIC_ACQUIRE, __HIP_MEMORY_SCOPE_AGENT)
           < target) {
      __builtin_amdgcn_s_sleep(1);
    }
    __threadfence();  // acquire: invalidate stale WGP$ lines
  }
  __syncthreads();
}

// Grid: REC_NWG blocks x 1024 threads (32 waves of wave32).
// WG `b` owns output dims [b*32, b*32+32) of all three gates, so gating is
// fused into stage 1.  Wave w covers K-chunk [w*32, w*32+32); partials are
// tree-reduced across the 32 waves in LDS.
__global__ __launch_bounds__(1024)
void ts_tree_recurrence(const float*  __restrict__ iou_x,   // [N, 3D]
                        const float*  __restrict__ fx,      // [N+1, D] (row N = 0)
                        const int*    __restrict__ parent,  // [N]
                        const __bf16* __restrict__ Wiouh,   // [D, 3D]
                        const float*  __restrict__ biouh,   // [3D]
                        const __bf16* __restrict__ Wfh,     // [D, D]
                        const float*  __restrict__ bfh,     // [D]
                        float* __restrict__ sum_h,          // [N+1, D] (zeroed)
                        float* __restrict__ sum_fc,         // [N+1, D] (zeroed)
                        float* __restrict__ c_buf,          // [D]
                        float* __restrict__ h_out,          // [N, D] == d_out
                        unsigned* __restrict__ bar_cnt) {   // zeroed
  __shared__ float lds_h[TD];
  __shared__ float red_i[TD];
  __shared__ float red_o[TD];
  __shared__ float red_u[TD];

  const int tid   = threadIdx.x;
  const int w     = tid >> 5;
  const int lane  = tid & 31;
  const int jbase = blockIdx.x * 32;
  const int j     = jbase + lane;
  unsigned barNum = 0;

  for (int t = 0; t < TN; ++t) {
    // ---- stage 1: iou = iou_x[t] + sum_h[t] @ Wiouh + biouh; c, h ----
    lds_h[tid] = sum_h[(size_t)t * TD + tid];
    __syncthreads();

    float ai = 0.f, ao = 0.f, au = 0.f;
    const __bf16* wp = Wiouh + (size_t)(w * 32) * TD3 + j;
#pragma unroll 4
    for (int kk = 0; kk < 32; ++kk) {
      float hv = lds_h[w * 32 + kk];
      ai += hv * (float)wp[0];
      ao += hv * (float)wp[TD];
      au += hv * (float)wp[2 * TD];
      wp += TD3;
    }
    red_i[tid] = ai; red_o[tid] = ao; red_u[tid] = au;
    for (int s = 16; s >= 1; s >>= 1) {
      __syncthreads();
      if (w < s) {
        red_i[tid] += red_i[tid + s * 32];
        red_o[tid] += red_o[tid + s * 32];
        red_u[tid] += red_u[tid + s * 32];
      }
    }
    __syncthreads();
    if (w == 0) {
      const float* ix = iou_x + (size_t)t * TD3;
      float iv = red_i[lane] + ix[j]          + biouh[j];
      float ov = red_o[lane] + ix[TD + j]     + biouh[TD + j];
      float uv = red_u[lane] + ix[2 * TD + j] + biouh[2 * TD + j];
      float c  = ts_sigmoid(iv) * tanhf(uv) + sum_fc[(size_t)t * TD + j];
      float h  = ts_sigmoid(ov) * tanhf(c);
      c_buf[j] = c;
      h_out[(size_t)t * TD + j] = h;
    }
    ++barNum;
    ts_grid_barrier(bar_cnt, barNum * REC_NWG);

    // ---- stage 2: f = sigmoid(h @ Wfh + bfh + fx[parent]); scatter ----
    lds_h[tid] = h_out[(size_t)t * TD + tid];
    __syncthreads();

    float af = 0.f;
    const __bf16* wf = Wfh + (size_t)(w * 32) * TD + j;
#pragma unroll 4
    for (int kk = 0; kk < 32; ++kk) {
      af += lds_h[w * 32 + kk] * (float)wf[0];
      wf += TD;
    }
    red_i[tid] = af;
    for (int s = 16; s >= 1; s >>= 1) {
      __syncthreads();
      if (w < s) red_i[tid] += red_i[tid + s * 32];
    }
    __syncthreads();
    if (w == 0) {
      int   p = parent[t];                          // p in [1, N]; p > t
      float f = ts_sigmoid(red_i[lane] + bfh[j] + fx[(size_t)p * TD + j]);
      // unique (p, j) writer this step; steps serialized by barrier
      sum_h [(size_t)p * TD + j] += lds_h[j];
      sum_fc[(size_t)p * TD + j] += f * c_buf[j];
    }
    ++barNum;
    ts_grid_barrier(bar_cnt, barNum * REC_NWG);
  }
}

// ------------------------------- launcher ----------------------------------

extern "C" void kernel_launch(void* const* d_in, const int* in_sizes, int n_in,
                              void* d_out, int out_size, void* d_ws, size_t ws_size,
                              hipStream_t stream) {
  (void)in_sizes; (void)n_in; (void)out_size; (void)ws_size;

  const float* inputs = (const float*)d_in[0];   // [N, D_IN]
  const int*   parent = (const int*)  d_in[1];   // [N]
  const float* Wioux  = (const float*)d_in[2];   // [D_IN, 3D]
  const float* bioux  = (const float*)d_in[3];   // [3D]
  const float* Wiouh  = (const float*)d_in[4];   // [D, 3D]
  const float* biouh  = (const float*)d_in[5];   // [3D]
  const float* Wfx    = (const float*)d_in[6];   // [D_IN, D]
  const float* bfx    = (const float*)d_in[7];   // [D]
  const float* Wfh    = (const float*)d_in[8];   // [D, D]
  const float* bfh    = (const float*)d_in[9];   // [D]
  float*       h_out  = (float*)d_out;           // [N, D]

  // ---- workspace layout (bytes, 256-aligned blocks) ----
  char* ws = (char*)d_ws;
  size_t off = 0;
  auto take = [&](size_t bytes) { char* p = ws + off; off = (off + bytes + 255) & ~(size_t)255; return p; };

  __bf16* x_bf      = (__bf16*)take((size_t)TN * TDIN * 2);       // inputs bf16
  __bf16* WiouxT_bf = (__bf16*)take((size_t)TD3 * TDIN * 2);      // Wioux^T bf16
  __bf16* WfxT_bf   = (__bf16*)take((size_t)TD  * TDIN * 2);      // Wfx^T bf16
  __bf16* Wiouh_bf  = (__bf16*)take((size_t)TDIN * TD3 * 2);      // Wiouh bf16 (row-major)
  __bf16* Wfh_bf    = (__bf16*)take((size_t)TD * TD * 2);         // Wfh bf16 (row-major)
  float*  iou_x     = (float*) take((size_t)TN * TD3 * 4);        // [N, 3D]
  // zeroed-every-call contiguous region starts here:
  float*  fx        = (float*) take((size_t)(TN + 1) * TD * 4);   // [N+1, D]
  float*  sum_h     = (float*) take((size_t)(TN + 1) * TD * 4);
  float*  sum_fc    = (float*) take((size_t)(TN + 1) * TD * 4);
  float*  c_buf     = (float*) take((size_t)TD * 4);
  unsigned* bar_cnt = (unsigned*)take(256);

  // ---- 1) precision conversion + weight transposes (bf16) ----
  {
    size_t n = (size_t)TN * TDIN;
    ts_f32_to_bf16<<<dim3((unsigned)((n + 255) / 256)), dim3(256), 0, stream>>>(inputs, x_bf, n);
  }
  {
    size_t n = (size_t)TDIN * TD3;
    ts_f32_to_bf16<<<dim3((unsigned)((n + 255) / 256)), dim3(256), 0, stream>>>(Wiouh, Wiouh_bf, n);
  }
  {
    size_t n = (size_t)TD * TD;
    ts_f32_to_bf16<<<dim3((unsigned)((n + 255) / 256)), dim3(256), 0, stream>>>(Wfh, Wfh_bf, n);
  }
  {
    size_t n = (size_t)TDIN * TD3;
    ts_f32_to_bf16_T<<<dim3((unsigned)((n + 255) / 256)), dim3(256), 0, stream>>>(Wioux, WiouxT_bf, TDIN, TD3);
  }
  {
    size_t n = (size_t)TDIN * TD;
    ts_f32_to_bf16_T<<<dim3((unsigned)((n + 255) / 256)), dim3(256), 0, stream>>>(Wfx, WfxT_bf, TDIN, TD);
  }

  // ---- 2) zero fx / sum_h / sum_fc / c_buf / barrier counter ----
  {
    size_t nz = (size_t)(TN + 1) * TD * 3 + TD + 64;  // contiguous region
    ts_zero_f32<<<dim3((unsigned)((nz + 255) / 256)), dim3(256), 0, stream>>>(fx, nz);
  }

  // ---- 3) batched input projections via WMMA ----
  ts_gemm_bf16_wmma<<<dim3(TD3 / 64, TN / 32), dim3(256), 0, stream>>>(
      x_bf, WiouxT_bf, bioux, iou_x, TN, TD3, TDIN);
  ts_gemm_bf16_wmma<<<dim3(TD / 64, TN / 32), dim3(256), 0, stream>>>(
      x_bf, WfxT_bf, bfx, fx, TN, TD, TDIN);   // rows 0..N-1; row N stays 0

  // ---- 4) sequential tree recurrence (persistent grid) ----
  ts_tree_recurrence<<<dim3(REC_NWG), dim3(1024), 0, stream>>>(
      iou_x, fx, parent, Wiouh_bf, biouh, Wfh_bf, bfh,
      sum_h, sum_fc, c_buf, h_out, bar_cnt);
}
